// FileEncoder_47888885350560
// MI455X (gfx1250) — matrix operand order repaired and checked by reading, not compiled, not run on hardware
//
#include <hip/hip_runtime.h>
#include <hip/hip_bf16.h>

// ---------------------------------------------------------------------------
// FileEncoder on MI455X (gfx1250): 2-layer transformer encoder.
// One workgroup (8 wave32) per sequence row; all GEMMs on
// v_wmma_f32_16x16x32_f16. Weights are repacked once into WMMA-B
// fragment-major tiles and streamed to LDS with the Tensor Data Mover
// (tensor_load_to_lds, TENSORcnt), double-buffered against compute.
// ---------------------------------------------------------------------------

#define NUM_LAYERS 2
#define NHEAD      4
#define HD         64
#define DIM        256
#define DFF        1024
#define NS_        2048
#define SEQ        64
#define NH_        512
#define NUM_RANKS  4

typedef __attribute__((ext_vector_type(16))) _Float16     v16h;
typedef __attribute__((ext_vector_type(8)))  float        v8f;
typedef __attribute__((ext_vector_type(4)))  unsigned int v4u;
typedef __attribute__((ext_vector_type(8)))  int          v8i_;
typedef __attribute__((ext_vector_type(4)))  int          v4i_;

// LDS strides (padded to dodge bank conflicts)
#define LDH    272   // halves per row for f16 [64 x 256] tiles
#define LDSC   68    // floats per row for the [64 x 64] score tile

// WMMA-B packed tile: 32 lanes x 16 halves = 512 halves = 1024 bytes
#define TILE_H       512
#define PANEL_TILES  16          // one K-step panel covers 16 N-tiles (256 cols)
#define PANEL_HALVES (PANEL_TILES * TILE_H)   // 8192 halves
#define PANEL_BYTES  (PANEL_HALVES * 2)       // 16 KB

// packed f16 weight offsets (in halves) inside d_ws
#define WQ_OFF      0
#define WK_OFF      131072
#define WV_OFF      262144
#define WO_OFF      393216
#define W1_OFF      524288
#define W2_OFF      1048576
#define W_TOTAL_H   1572864
#define ATT_LSTRIDE 65536          // 256*256 per layer
#define FFN_LSTRIDE 262144         // 256*1024 per layer

__device__ __forceinline__ int lane_id() { return (int)(threadIdx.x & 31u); }
__device__ __forceinline__ int wave_id() { return (int)(threadIdx.x >> 5); }

__device__ __forceinline__ float gelu_f(float x) {
  return 0.5f * x * (1.0f + tanhf(0.7978845608f * (x + 0.044715f * x * x * x)));
}

// ---- Tensor Data Mover: contiguous 1-D global -> LDS DMA (D# per ISA 8.3-8.5)
__device__ __forceinline__ void tdm_load_1d(_Float16* lds_dst, const _Float16* gsrc,
                                            unsigned bytes) {
#if __has_builtin(__builtin_amdgcn_tensor_load_to_lds)
  const unsigned long long ga = (unsigned long long)(uintptr_t)gsrc;
  const unsigned lds_off = (unsigned)(uintptr_t)lds_dst;  // LDS aperture: low 32 bits
  const unsigned ne8 = bytes >> 3;                        // 8-byte elements
  v4u g0;
  g0.x = 1u;                                              // count=1, user descriptor
  g0.y = lds_off;                                         // lds_addr
  g0.z = (unsigned)ga;                                    // global_addr[31:0]
  g0.w = (unsigned)((ga >> 32) & 0x01ffffffu) | (2u << 30); // global_addr[56:32], type=2
  v8i_ g1;
  g1[0] = (int)(3u << 16);                                // workgroup_mask=0, data_size=8B
  g1[1] = (int)((ne8 & 0xffffu) << 16);                   // tensor_dim0[15:0]
  g1[2] = (int)(((ne8 >> 16) & 0xffffu) | (1u << 16));    // tensor_dim0[31:16], tensor_dim1=1
  g1[3] = (int)((ne8 & 0xffffu) << 16);                   // tile_dim0 = ne8
  g1[4] = 0;                                              // tile_dim1/2 unused (1-D)
  g1[5] = (int)ne8;                                       // tensor_dim0_stride[31:0]
  g1[6] = 0;
  g1[7] = 0;
  v4i_ gz;  gz[0] = gz[1] = gz[2] = gz[3] = 0;            // groups 2/3 unused (<=2D)
#if defined(__clang_major__) && (__clang_major__ >= 23)
  v8i_ gz8; gz8[0]=gz8[1]=gz8[2]=gz8[3]=gz8[4]=gz8[5]=gz8[6]=gz8[7]=0;
  __builtin_amdgcn_tensor_load_to_lds(g0, g1, gz, gz, gz8, 0);
#else
  __builtin_amdgcn_tensor_load_to_lds(g0, g1, gz, gz, 0);
#endif
#else
  // fallback: lane-cooperative copy by the issuing wave
  const int l = lane_id();
  const v16h* s = (const v16h*)gsrc;
  v16h* d = (v16h*)lds_dst;
  for (unsigned i = (unsigned)l; i < (bytes >> 5); i += 32) d[i] = s[i];
#endif
}

__device__ __forceinline__ void wait_tensor0() {
#if __has_builtin(__builtin_amdgcn_s_wait_tensorcnt)
  __builtin_amdgcn_s_wait_tensorcnt((short)0);
#endif
}

// ---- WMMA fragment builders (wave32, contiguous k = 16*(lane/16) + i in A and B) -------------------

// A fragment 16x32 f16 from LDS row-major [rows x ld].
__device__ __forceinline__ v16h frag_a_f16(const _Float16* p, int ld, int row0, int k0) {
  const int l   = lane_id();
  const int m   = l & 15;
  const int hi8 = (l >> 4) << 4;
  const _Float16* r = p + (row0 + m) * ld + k0;
  v16h a;
#pragma unroll
  for (int i = 0; i < 16; ++i) {
    const int k = i + hi8;
    a[i] = r[k];
  }
  return a;
}

// A fragment from f32 LDS (attention probabilities), converted to f16.
__device__ __forceinline__ v16h frag_a_f32(const float* p, int ld, int row0, int k0) {
  const int l   = lane_id();
  const int m   = l & 15;
  const int hi8 = (l >> 4) << 4;
  const float* r = p + (row0 + m) * ld + k0;
  v16h a;
#pragma unroll
  for (int i = 0; i < 16; ++i) {
    const int k = i + hi8;
    a[i] = (_Float16)r[k];
  }
  return a;
}

// B fragment from a packed panel in LDS: one 32-byte vector load per lane.
__device__ __forceinline__ v16h frag_b_panel(const _Float16* pan, int nt_tile) {
  const v16h* t = (const v16h*)(pan + nt_tile * TILE_H);
  return t[lane_id()];
}

// B fragment 32x16 from a row-major [K x N] LDS source (V tile for probs@V).
__device__ __forceinline__ v16h frag_b_rm(const _Float16* p, int ld, int k0, int n0) {
  const int l    = lane_id();
  const int n    = l & 15;
  const int hi16 = (l >> 4) << 4;
  v16h b;
#pragma unroll
  for (int i = 0; i < 16; ++i) b[i] = p[(k0 + hi16 + i) * ld + n0 + n];
  return b;
}

// B fragment with on-the-fly transpose: B[k][n] = p[n][k] (K^T for scores).
__device__ __forceinline__ v16h frag_b_tr(const _Float16* p, int ld, int k0, int n0) {
  const int l    = lane_id();
  const int n    = l & 15;
  const int hi16 = (l >> 4) << 4;
  v16h b;
#pragma unroll
  for (int i = 0; i < 16; ++i) b[i] = p[(n0 + n) * ld + k0 + hi16 + i];
  return b;
}

__device__ __forceinline__ void store_c_f16(_Float16* p, int ld, int row0, int col0, v8f c) {
  const int l  = lane_id();
  const int n  = l & 15;
  const int mb = (l >> 4) << 3;
#pragma unroll
  for (int r = 0; r < 8; ++r)
    p[(row0 + mb + r) * ld + col0 + n] = (_Float16)c[r];
}

__device__ __forceinline__ void store_c_f32s(float* p, int ld, int row0, int col0, v8f c, float s) {
  const int l  = lane_id();
  const int n  = l & 15;
  const int mb = (l >> 4) << 3;
#pragma unroll
  for (int r = 0; r < 8; ++r)
    p[(row0 + mb + r) * ld + col0 + n] = c[r] * s;
}

__device__ __forceinline__ v8f wmma16(v16h a, v16h b, v8f c) {
  return __builtin_amdgcn_wmma_f32_16x16x32_f16(false, a, false, b, (short)0, c, false, false);
}

// ---------------------------------------------------------------------------
// Panel GEMM: acc[8] += A[64 x 32*n_ktiles] * B (packed tiles). Each wave owns
// M-tile (wave&3) and 8 N-tiles; one A fragment is reused across all 8 WMMAs.
// Weight K-panels (16 KB) are TDM-streamed into a double-buffered LDS panel.
// ---------------------------------------------------------------------------
__device__ void gemm_panel(v8f acc[8], const _Float16* __restrict__ A, int lda,
                           const _Float16* __restrict__ Bp, int kt_stride_tiles,
                           int n_ktiles, _Float16* panel) {
  const int wave = wave_id();
  const int mt   = (wave & 3) << 4;
  if (wave == 0) tdm_load_1d(panel, Bp, PANEL_BYTES);
  for (int kt = 0; kt < n_ktiles; ++kt) {
    if (wave == 0) wait_tensor0();
    __syncthreads();                              // panel[kt&1] visible to all waves
    if (wave == 0 && kt + 1 < n_ktiles)
      tdm_load_1d(panel + ((kt + 1) & 1) * PANEL_HALVES,
                  Bp + (size_t)(kt + 1) * kt_stride_tiles * TILE_H, PANEL_BYTES);
    const _Float16* pan = panel + (kt & 1) * PANEL_HALVES;
    const v16h a = frag_a_f16(A, lda, mt, kt << 5);
#pragma unroll
    for (int i = 0; i < 8; ++i) {
      const v16h b = frag_b_panel(pan, (wave >> 2) + (i << 1));
      acc[i] = wmma16(a, b, acc[i]);
    }
  }
  __syncthreads();                                // safe to re-stage panel next call
}

__device__ __forceinline__ void store_tiles(const v8f acc[8], _Float16* D, int ldd,
                                            bool do_gelu) {
  const int wave = wave_id();
  const int mt   = (wave & 3) << 4;
#pragma unroll
  for (int i = 0; i < 8; ++i) {
    const int nt = ((wave >> 2) + (i << 1)) << 4;
    v8f c = acc[i];
    if (do_gelu) {
#pragma unroll
      for (int r = 0; r < 8; ++r) c[r] = gelu_f(c[r]);
    }
    store_c_f16(D, ldd, mt, nt, c);
  }
}

// Fused residual + LayerNorm over DIM=256, thread-per-row.
__device__ void ln_residual(_Float16* x, const _Float16* y,
                            const float* __restrict__ s, const float* __restrict__ b) {
  const int r = (int)threadIdx.x;
  if (r < SEQ) {
    const _Float16* xr = x + r * LDH;
    const _Float16* yr = y + r * LDH;
    float m = 0.f;
    for (int c = 0; c < DIM; ++c) m += (float)xr[c] + (float)yr[c];
    m *= (1.0f / DIM);
    float var = 0.f;
    for (int c = 0; c < DIM; ++c) {
      float d = (float)xr[c] + (float)yr[c] - m;
      var += d * d;
    }
    const float rs = rsqrtf(var * (1.0f / DIM) + 1e-5f);
    for (int c = 0; c < DIM; ++c) {
      float v = ((float)xr[c] + (float)yr[c] - m) * rs * s[c] + b[c];
      x[r * LDH + c] = (_Float16)v;
    }
  }
}

// ---------------------------------------------------------------------------
// One block = one sequence row.
// ---------------------------------------------------------------------------
__global__ void term_encoder_kernel(
    const int* __restrict__ tokens, const unsigned char* __restrict__ refm,
    const unsigned char* __restrict__ padm, const int* __restrict__ sortv, int rank_sel,
    const float* __restrict__ reprs,
    const float* __restrict__ emb0, const float* __restrict__ emb1,
    const _Float16* __restrict__ w,
    const float* __restrict__ ln1s, const float* __restrict__ ln1b,
    const float* __restrict__ ln2s, const float* __restrict__ ln2b,
    float* __restrict__ out_cls) {
  const int row = (int)blockIdx.x;
  if (rank_sel >= 0 && sortv[row] != rank_sel) return;

  extern __shared__ unsigned char smem_raw[];
  _Float16* x16   = (_Float16*)smem_raw;        // [64][LDH] residual stream
  _Float16* q16   = x16 + SEQ * LDH;            // Q (reused: Wo out)
  _Float16* k16   = q16 + SEQ * LDH;            // K (reused: FFN hidden chunk)
  _Float16* v16   = k16 + SEQ * LDH;            // V (reused: FFN out)
  _Float16* t16   = v16 + SEQ * LDH;            // attention output
  _Float16* panel = t16 + SEQ * LDH;            // [2][PANEL_HALVES] TDM staging
  float*    sc    = (float*)(panel + 2 * PANEL_HALVES);  // [64][LDSC]

  const int tid  = (int)threadIdx.x;
  const int wave = tid >> 5;

  // ---- token embedding + injection of previously computed tree reprs ------
  for (int idx = tid; idx < SEQ * DIM; idx += 256) {
    const int t = idx >> 8, d = idx & 255;
    const int t0 = tokens[(row * SEQ + t) * 2 + 0];
    const int t1 = tokens[(row * SEQ + t) * 2 + 1];
    const float val = refm[row * SEQ + t] ? reprs[t1 * DIM + d]
                                          : (emb0[t0 * DIM + d] + emb1[t1 * DIM + d]);
    x16[t * LDH + d] = (_Float16)val;
  }
  __syncthreads();

  for (int l = 0; l < NUM_LAYERS; ++l) {
    const _Float16* Wq = w + WQ_OFF + l * ATT_LSTRIDE;
    const _Float16* Wk = w + WK_OFF + l * ATT_LSTRIDE;
    const _Float16* Wv = w + WV_OFF + l * ATT_LSTRIDE;
    const _Float16* Wo = w + WO_OFF + l * ATT_LSTRIDE;
    const _Float16* W1 = w + W1_OFF + l * FFN_LSTRIDE;
    const _Float16* W2 = w + W2_OFF + l * FFN_LSTRIDE;

    // ---- Q,K,V projections (TDM-streamed weight panels) --------------------
    {
      v8f a0[8] = {};
      gemm_panel(a0, x16, LDH, Wq, PANEL_TILES, 8, panel);
      store_tiles(a0, q16, LDH, false);
      v8f a1[8] = {};
      gemm_panel(a1, x16, LDH, Wk, PANEL_TILES, 8, panel);
      store_tiles(a1, k16, LDH, false);
      v8f a2[8] = {};
      gemm_panel(a2, x16, LDH, Wv, PANEL_TILES, 8, panel);
      store_tiles(a2, v16, LDH, false);
    }
    __syncthreads();

    // ---- attention, one head at a time ------------------------------------
    for (int h = 0; h < NHEAD; ++h) {
      const int ho = h * HD;
      // scores = (Q_h @ K_h^T) / sqrt(HD): 16 tiles, 2 per wave
      for (int t = wave; t < 16; t += 8) {
        const int mt = (t & 3) << 4, nt = (t >> 2) << 4;
        v8f acc = {};
        for (int k0 = 0; k0 < HD; k0 += 32) {
          v16h a = frag_a_f16(q16, LDH, mt, ho + k0);
          v16h b = frag_b_tr(k16, LDH, ho + k0, nt);
          acc = wmma16(a, b, acc);
        }
        store_c_f32s(sc, LDSC, mt, nt, acc, 0.125f);
      }
      __syncthreads();
      // masked softmax, thread-per-query-row
      if (tid < SEQ) {
        float mx = -3.0e38f;
        for (int c = 0; c < SEQ; ++c) {
          const float v = padm[row * SEQ + c] ? sc[tid * LDSC + c] : -1.0e9f;
          sc[tid * LDSC + c] = v;
          mx = fmaxf(mx, v);
        }
        float sum = 0.f;
        for (int c = 0; c < SEQ; ++c) {
          const float e = __expf(sc[tid * LDSC + c] - mx);
          sc[tid * LDSC + c] = e;
          sum += e;
        }
        const float inv = 1.0f / sum;
        for (int c = 0; c < SEQ; ++c) sc[tid * LDSC + c] *= inv;
      }
      __syncthreads();
      // O_h = probs @ V_h -> t16 columns [ho, ho+HD)
      for (int t = wave; t < 16; t += 8) {
        const int mt = (t & 3) << 4, nt = (t >> 2) << 4;
        v8f acc = {};
        for (int k0 = 0; k0 < SEQ; k0 += 32) {
          v16h a = frag_a_f32(sc, LDSC, mt, k0);
          v16h b = frag_b_rm(v16, LDH, k0, ho + nt);
          acc = wmma16(a, b, acc);
        }
        store_c_f16(t16, LDH, mt, ho + nt, acc);
      }
      __syncthreads();
    }

    // ---- output projection + residual LN ----------------------------------
    {
      v8f ao[8] = {};
      gemm_panel(ao, t16, LDH, Wo, PANEL_TILES, 8, panel);
      store_tiles(ao, q16, LDH, false);
    }
    __syncthreads();
    ln_residual(x16, q16, ln1s + l * DIM, ln1b + l * DIM);
    __syncthreads();

    // ---- FFN: 4 hidden-column chunks of 256; y accumulates in registers ----
    {
      v8f accY[8] = {};
      for (int c = 0; c < 4; ++c) {
        v8f accH[8] = {};
        // hidden chunk = gelu(x @ W1[:, c*256 : c*256+256]) -> k16 (reused)
        gemm_panel(accH, x16, LDH, W1 + (size_t)c * PANEL_TILES * TILE_H,
                   DFF >> 4 /* 64 tiles per kt row */, 8, panel);
        store_tiles(accH, k16, LDH, true);
        __syncthreads();
        // y += h_chunk @ W2[c*256 : c*256+256, :]
        gemm_panel(accY, k16, LDH, W2 + (size_t)c * 8 * PANEL_TILES * TILE_H,
                   PANEL_TILES, 8, panel);
      }
      store_tiles(accY, v16, LDH, false);
    }
    __syncthreads();
    ln_residual(x16, v16, ln2s + l * DIM, ln2b + l * DIM);
    __syncthreads();
  }

  // CLS (position 0) representation
  if (tid < DIM) out_cls[row * DIM + tid] = (float)x16[0 * LDH + tid];
}

// ---------------------------------------------------------------------------
// Repack W [K x N] f32 row-major -> f16 WMMA-B fragment-major tiles:
// dst[(kt*(N/16)+nt)*512 + lane*16 + i] = W[(kt*32 + (lane>>4)*16 + i)*N + nt*16 + (lane&15)]
// ---------------------------------------------------------------------------
__global__ void pack_b_kernel(const float* __restrict__ src, _Float16* __restrict__ dst,
                              int K, int N) {
  const int idx = (int)(blockIdx.x * blockDim.x + threadIdx.x);
  if (idx >= K * N) return;
  const int t = idx >> 9;
  const int r = idx & 511;
  const int l = r >> 4, i = r & 15;
  const int ntiles = N >> 4;
  const int kt = t / ntiles, nt = t % ntiles;
  const int k = kt * 32 + ((l >> 4) << 4) + i;
  const int n = nt * 16 + (l & 15);
  dst[idx] = (_Float16)src[k * N + n];
}

__global__ void zero_f32_kernel(float* __restrict__ p, int n) {
  const int i = (int)(blockIdx.x * blockDim.x + threadIdx.x);
  if (i < n) p[i] = 0.0f;
}

__global__ void scatter_cls_kernel(const float* __restrict__ tmp, const int* __restrict__ sortv,
                                   int r, float* __restrict__ reprs, int n) {
  const int i = (int)(blockIdx.x * blockDim.x + threadIdx.x);
  if (i < n && sortv[i >> 8] == r) reprs[i] = tmp[i];
}

// ---------------------------------------------------------------------------
extern "C" void kernel_launch(void* const* d_in, const int* in_sizes, int n_in,
                              void* d_out, int out_size, void* d_ws, size_t ws_size,
                              hipStream_t stream) {
  (void)in_sizes; (void)n_in; (void)out_size; (void)ws_size;

  const int*           tokens_scope = (const int*)d_in[0];
  const int*           scope_sort   = (const int*)d_in[1];
  const unsigned char* pad_scope    = (const unsigned char*)d_in[2];
  const unsigned char* ref_scope    = (const unsigned char*)d_in[3];
  const int*           tokens_hole  = (const int*)d_in[4];
  const unsigned char* pad_hole     = (const unsigned char*)d_in[5];
  const unsigned char* ref_hole     = (const unsigned char*)d_in[6];
  const float*         emb0         = (const float*)d_in[7];
  const float*         emb1         = (const float*)d_in[8];
  const float*         Wq           = (const float*)d_in[9];
  const float*         Wk           = (const float*)d_in[10];
  const float*         Wv           = (const float*)d_in[11];
  const float*         Wo           = (const float*)d_in[12];
  const float*         W1           = (const float*)d_in[13];
  const float*         W2           = (const float*)d_in[14];
  const float*         ln1s         = (const float*)d_in[15];
  const float*         ln1b         = (const float*)d_in[16];
  const float*         ln2s         = (const float*)d_in[17];
  const float*         ln2b         = (const float*)d_in[18];

  float* scope_reprs = (float*)d_out;               // [NS, DIM] live repr table
  float* hole_reprs  = scope_reprs + NS_ * DIM;     // [NH, DIM]

  _Float16* w16     = (_Float16*)d_ws;
  float*    tmp_cls = (float*)((char*)d_ws + (size_t)W_TOTAL_H * sizeof(_Float16));

  zero_f32_kernel<<<(NS_ * DIM + 255) / 256, 256, 0, stream>>>(scope_reprs, NS_ * DIM);

  // one-time f32 -> f16 repack into WMMA-B fragment-major tiles (L2-resident)
  for (int l = 0; l < NUM_LAYERS; ++l) {
    pack_b_kernel<<<(ATT_LSTRIDE + 255) / 256, 256, 0, stream>>>(
        Wq + (size_t)l * ATT_LSTRIDE, w16 + WQ_OFF + (size_t)l * ATT_LSTRIDE, DIM, DIM);
    pack_b_kernel<<<(ATT_LSTRIDE + 255) / 256, 256, 0, stream>>>(
        Wk + (size_t)l * ATT_LSTRIDE, w16 + WK_OFF + (size_t)l * ATT_LSTRIDE, DIM, DIM);
    pack_b_kernel<<<(ATT_LSTRIDE + 255) / 256, 256, 0, stream>>>(
        Wv + (size_t)l * ATT_LSTRIDE, w16 + WV_OFF + (size_t)l * ATT_LSTRIDE, DIM, DIM);
    pack_b_kernel<<<(ATT_LSTRIDE + 255) / 256, 256, 0, stream>>>(
        Wo + (size_t)l * ATT_LSTRIDE, w16 + WO_OFF + (size_t)l * ATT_LSTRIDE, DIM, DIM);
    pack_b_kernel<<<(FFN_LSTRIDE + 255) / 256, 256, 0, stream>>>(
        W1 + (size_t)l * FFN_LSTRIDE, w16 + W1_OFF + (size_t)l * FFN_LSTRIDE, DIM, DFF);
    pack_b_kernel<<<(FFN_LSTRIDE + 255) / 256, 256, 0, stream>>>(
        W2 + (size_t)l * FFN_LSTRIDE, w16 + W2_OFF + (size_t)l * FFN_LSTRIDE, DFF, DIM);
  }

  const size_t smem = (size_t)(5 * SEQ * LDH) * sizeof(_Float16)   // activations
                    + (size_t)(2 * PANEL_HALVES) * sizeof(_Float16) // TDM panels
                    + (size_t)(SEQ * LDSC) * sizeof(float);         // scores (~219 KB)

  // topological rank loop: only rows of rank r are encoded, then committed
  for (int r = 0; r < NUM_RANKS; ++r) {
    term_encoder_kernel<<<NS_, 256, smem, stream>>>(
        tokens_scope, ref_scope, pad_scope, scope_sort, r, scope_reprs,
        emb0, emb1, w16, ln1s, ln1b, ln2s, ln2b, tmp_cls);
    scatter_cls_kernel<<<(NS_ * DIM + 255) / 256, 256, 0, stream>>>(
        tmp_cls, scope_sort, r, scope_reprs, NS_ * DIM);
  }

  // hole sequences: encode all rows against the finished repr table
  term_encoder_kernel<<<NH_, 256, smem, stream>>>(
      tokens_hole, ref_hole, pad_hole, nullptr, -1, scope_reprs,
      emb0, emb1, w16, ln1s, ln1b, ln2s, ln2b, hole_reprs);
}
